// GTAttention_59949153517851
// MI455X (gfx1250) — matrix operand order repaired and checked by reading, not compiled, not run on hardware
//
#include <hip/hip_runtime.h>

typedef __attribute__((ext_vector_type(16))) _Float16 v16h;
typedef __attribute__((ext_vector_type(8)))  _Float16 v8h;
typedef __attribute__((ext_vector_type(4)))  _Float16 v4h;
typedef __attribute__((ext_vector_type(8)))  float    v8f;
typedef __attribute__((ext_vector_type(4)))  float    v4f;

#define EMB   768
#define NTOK  16384     // B * L
#define NH    24
#define HD    32
#define LSEQ  1024

// ---------------------------------------------------------------------------
// WMMA helper: D = A(16x32 f16) * B(32x16 f16) + C(16x16 f32), wave32
// ---------------------------------------------------------------------------
__device__ __forceinline__ v8f wmma_f16f32(v16h a, v16h b, v8f c) {
  return __builtin_amdgcn_wmma_f32_16x16x32_f16(
      /*neg_a=*/false, a, /*neg_b=*/false, b,
      /*c_mod=*/(short)0, c, /*reuse_a=*/false, /*reuse_b=*/false);
}

// A fragment (16x32): source stored row-major [m][k] in LDS.
__device__ __forceinline__ v16h load_a_frag(const _Float16* sBase, int ld,
                                            int kOff, int lane) {
  const int rowM = lane & 15, half = lane >> 4;
  const _Float16* p = sBase + rowM * ld + kOff + half * 16;
  v8h lo = *(const v8h*)p;          // K = half*16 + 0..7
  v8h hi = *(const v8h*)(p + 8);    // K = half*16 + 8..15
  v16h a;
#pragma unroll
  for (int j = 0; j < 8; ++j) { a[j] = lo[j]; a[8 + j] = hi[j]; }
  return a;
}

// B fragment (32x16): source stored [n][k] in LDS.
__device__ __forceinline__ v16h load_b_frag_nk(const _Float16* sBase, int ld,
                                               int lane) {
  const int colN = lane & 15, half = lane >> 4;
  const _Float16* p = sBase + colN * ld + half * 16;
  v8h x0 = *(const v8h*)p;
  v8h x1 = *(const v8h*)(p + 8);
  v16h b;
#pragma unroll
  for (int j = 0; j < 8; ++j) { b[j] = x0[j]; b[8 + j] = x1[j]; }
  return b;
}

// B fragment (32x16): source stored [k][n] in LDS (e.g. V tile).
__device__ __forceinline__ v16h load_b_frag_kn(const _Float16* sBase, int ld,
                                               int lane) {
  const int colN = lane & 15, half = lane >> 4;
  v16h b;
#pragma unroll
  for (int j = 0; j < 16; ++j) b[j] = sBase[(half * 16 + j) * ld + colN];
  return b;
}

// ---------------------------------------------------------------------------
// Staging helpers (LDS row stride 40 f16 = 80 B; bases 16 B aligned).
// ---------------------------------------------------------------------------
// 128 rows x 32 cols fp32 -> f16 LDS; branch-free: 4 float4 per thread.
__device__ __forceinline__ void stage128_f32cvt(_Float16* s, const float* g,
                                                size_t gStride, int tid) {
#pragma unroll
  for (int k = 0; k < 4; ++k) {
    const int i = tid + k * 256;            // 0..1023
    const int r = i >> 3, c4 = (i & 7) * 4;
    v4f x = *(const v4f*)(g + (size_t)r * gStride + c4);
    v4h h;
    h[0] = (_Float16)x[0]; h[1] = (_Float16)x[1];
    h[2] = (_Float16)x[2]; h[3] = (_Float16)x[3];
    *(v4h*)(s + r * 40 + c4) = h;
  }
}

__device__ __forceinline__ unsigned lds_addr32(const void* p) {
  return (unsigned)(size_t)p;  // generic LDS pointer: low 32 bits = LDS offset
}

__device__ __forceinline__ void async_b128(unsigned lds, const void* ga) {
  asm volatile("global_load_async_to_lds_b128 %0, %1, off"
               :: "v"(lds), "v"(ga) : "memory");
}

// 64 rows x 32 f16 tile: one async b128 per thread.
__device__ __forceinline__ void async_stage64_f16(unsigned ldsBase,
                                                  const _Float16* g,
                                                  size_t gStride, int tid) {
  const int r = tid >> 2, c = tid & 3;
  async_b128(ldsBase + (unsigned)(r * 80 + c * 16),
             (const void*)(g + (size_t)r * gStride + c * 8));
}

// 128 rows x 32 f16 tile: two async b128 per thread.
__device__ __forceinline__ void async_stage128_f16(unsigned ldsBase,
                                                   const _Float16* g,
                                                   size_t gStride, int tid) {
#pragma unroll
  for (int k = 0; k < 2; ++k) {
    const int i = tid + k * 256;
    const int r = i >> 2, c = i & 3;
    async_b128(ldsBase + (unsigned)(r * 80 + c * 16),
               (const void*)(g + (size_t)r * gStride + c * 8));
  }
}

template <int N>
__device__ __forceinline__ void wait_async() {
#if __has_builtin(__builtin_amdgcn_s_wait_asynccnt)
  __builtin_amdgcn_s_wait_asynccnt(N);
#else
  if constexpr (N == 0) asm volatile("s_wait_asynccnt 0" ::: "memory");
  else asm volatile("s_wait_asynccnt 2" ::: "memory");
#endif
}

// ---------------------------------------------------------------------------
// Kernel 1: qkv = x @ W_qkv^T + b, then SO(2) rotate pairs and scatter
// rotated Q/K/V in f16, layout (B, H, L, D).  Q pre-scaled by 1/sqrt(HD).
// Grid: (NTOK/128, 2304/128), block 256 (8 waves), 128x128 tile / block.
// Wave w -> M-tiles {(w>>1)*2, +1}, N-tiles {(w&1)*4 .. +3}: 8 WMMA / K-step.
// ---------------------------------------------------------------------------
__global__ __launch_bounds__(256) void qkv_rot_kernel(
    const float* __restrict__ x,     // (16384, 768)
    const float* __restrict__ Wqkv,  // (2304, 768)
    const float* __restrict__ bqkv,  // (2304)
    const float* __restrict__ matq,  // (1024, 16, 2, 2)
    const float* __restrict__ matk,
    const float* __restrict__ matv,
    _Float16* __restrict__ Qr,       // (B, H, L, D) f16
    _Float16* __restrict__ Kr,
    _Float16* __restrict__ Vr) {
  __shared__ __align__(16) _Float16 sA[128 * 40];
  __shared__ __align__(16) _Float16 sB[128 * 40];
  const int tid = threadIdx.x, wave = tid >> 5, lane = tid & 31;
  const int col = lane & 15, half = lane >> 4;
  const int mBlock = blockIdx.x * 128;  // token base
  const int nBlock = blockIdx.y * 128;  // qkv-column base
  const int waveM = wave >> 1;          // 0..3
  const int waveN = wave & 1;           // 0..1

  v8f acc[2][4] = {};
  for (int kt = 0; kt < EMB; kt += 32) {
    stage128_f32cvt(sA, x + (size_t)mBlock * EMB + kt, EMB, tid);
    stage128_f32cvt(sB, Wqkv + (size_t)nBlock * EMB + kt, EMB, tid);
    __syncthreads();
    v16h a0 = load_a_frag(sA + (waveM * 2 + 0) * 16 * 40, 40, 0, lane);
    v16h a1 = load_a_frag(sA + (waveM * 2 + 1) * 16 * 40, 40, 0, lane);
    v16h bf[4];
#pragma unroll
    for (int tn = 0; tn < 4; ++tn)
      bf[tn] = load_b_frag_nk(sB + (waveN * 4 + tn) * 16 * 40, 40, lane);
#pragma unroll
    for (int tn = 0; tn < 4; ++tn) {
      acc[0][tn] = wmma_f16f32(a0, bf[tn], acc[0][tn]);
      acc[1][tn] = wmma_f16f32(a1, bf[tn], acc[1][tn]);
    }
    __syncthreads();
  }

  // Epilogue: bias + SO(2) rotation. Even/odd head-dim partner columns sit in
  // adjacent lanes of the D fragment -> exchange with shfl_xor(1).
#pragma unroll
  for (int tm = 0; tm < 2; ++tm) {
#pragma unroll
    for (int tn = 0; tn < 4; ++tn) {
      const int n = nBlock + (waveN * 4 + tn) * 16 + col;  // 0..2303
      const int part = n / EMB;                            // 0=q 1=k 2=v
      const int e = n - part * EMB;
      const int hd = e >> 5, d = e & 31, f = d >> 1;
      const float* mat = (part == 0) ? matq : (part == 1) ? matk : matv;
      _Float16* dst    = (part == 0) ? Qr   : (part == 1) ? Kr   : Vr;
      const float postScale = (part == 0) ? 0.17677669529663687f : 1.0f;
      const float bias = bqkv[n];
#pragma unroll
      for (int v = 0; v < 8; ++v) {
        const int m = mBlock + (waveM * 2 + tm) * 16 + half * 8 + v;  // token
        const int l = m & (LSEQ - 1);
        const int bt = m >> 10;
        const float val = acc[tm][tn][v] + bias;
        const float other = __shfl_xor(val, 1);
        const float xe = (d & 1) ? other : val;   // x[2f]
        const float xo = (d & 1) ? val : other;   // x[2f+1]
        const float* mrow = mat + ((size_t)l * 16 + f) * 4 + (d & 1) * 2;
        const float y = (mrow[0] * xe + mrow[1] * xo) * postScale;
        dst[(((size_t)bt * NH + hd) * LSEQ + l) * HD + d] = (_Float16)y;
      }
    }
  }
}

// ---------------------------------------------------------------------------
// Kernel 2: flash attention per (b, h, 64-row q tile) + inverse rotation.
// K/V tiles double-buffered via GLOBAL_LOAD_ASYNC_TO_LDS_B128: chunk i+1 DMA
// overlaps chunk i compute; s_wait_asynccnt<=2 retires exactly chunk i.
// Grid: B*NH*(L/64) = 6144 blocks, block 256 (8 waves).
// ---------------------------------------------------------------------------
__global__ __launch_bounds__(256) void attn_kernel(
    const _Float16* __restrict__ Q, const _Float16* __restrict__ K,
    const _Float16* __restrict__ V, const float* __restrict__ mato,
    _Float16* __restrict__ outE)    // (B, L, 768) f16
{
  __shared__ __align__(16) _Float16 sQ[64 * 40];
  __shared__ __align__(16) _Float16 sK[2][64 * 40];
  __shared__ __align__(16) _Float16 sV[2][64 * 40];
  __shared__ __align__(16) _Float16 sP[64 * 72];
  __shared__ float sS[64 * 68];
  __shared__ float sO[64 * 36];
  __shared__ float sM[64], sSum[64], sCorr[64];

  const int tid = threadIdx.x, wave = tid >> 5, lane = tid & 31;
  const int col = lane & 15, half = lane >> 4;
  const int gid = blockIdx.x;
  const int qt = gid & 15;              // q tile (L/64 == 16)
  const int hd = (gid >> 4) % NH;       // head
  const int b  = gid / (16 * NH);       // batch
  const size_t hb = ((size_t)b * NH + hd) * LSEQ * HD;
  const int qRow0 = qt * 64;
  const int qi = wave >> 1, di = wave & 1;

  // Prologue: async Q tile (pre-scaled in kernel 1) + chunk 0 of K/V.
  async_stage64_f16(lds_addr32(sQ), Q + hb + (size_t)qRow0 * HD, HD, tid);
  async_stage64_f16(lds_addr32(&sK[0][0]), K + hb, HD, tid);
  async_stage64_f16(lds_addr32(&sV[0][0]), V + hb, HD, tid);
  if (tid < 64) { sM[tid] = -3.0e38f; sSum[tid] = 0.f; }
  v8f o = {};

  int ib = 0;
  for (int kt = 0; kt < LSEQ; kt += 64, ib ^= 1) {
    if (kt + 64 < LSEQ) {  // issue next chunk into the other buffer
      async_stage64_f16(lds_addr32(&sK[ib ^ 1][0]),
                        K + hb + (size_t)(kt + 64) * HD, HD, tid);
      async_stage64_f16(lds_addr32(&sV[ib ^ 1][0]),
                        V + hb + (size_t)(kt + 64) * HD, HD, tid);
      wait_async<2>();     // retire everything except the 2 just issued
    } else {
      wait_async<0>();
    }
    __syncthreads();       // chunk kt resident for all waves

    // S tile: wave computes rows qi*16, column tiles (di*2, di*2+1).
    {
      v16h a = load_a_frag(sQ + qi * 16 * 40, 40, 0, lane);
#pragma unroll
      for (int t = 0; t < 2; ++t) {
        const int nTile = di * 2 + t;
        v16h bf = load_b_frag_nk(&sK[ib][0] + nTile * 16 * 40, 40, lane);
        v8f sc = {};
        sc = wmma_f16f32(a, bf, sc);
#pragma unroll
        for (int v = 0; v < 8; ++v)
          sS[(qi * 16 + half * 8 + v) * 68 + nTile * 16 + col] = sc[v];
      }
    }
    __syncthreads();

    // Online softmax (fp32), one thread per q row.
    if (tid < 64) {
      const float* row = &sS[tid * 68];
      const float mOld = sM[tid];
      float mx = mOld;
      for (int j = 0; j < 64; ++j) mx = fmaxf(mx, row[j]);
      const float cf = __expf(mOld - mx);
      float sum = 0.f;
      for (int j = 0; j < 64; ++j) {
        const float p = __expf(row[j] - mx);
        sum += p;
        sP[tid * 72 + j] = (_Float16)p;
      }
      sM[tid] = mx;
      sSum[tid] = sSum[tid] * cf + sum;
      sCorr[tid] = cf;
    }
    __syncthreads();

    // O = O*corr + P @ V  (K chunk of 64 -> two K=32 WMMAs).
#pragma unroll
    for (int v = 0; v < 8; ++v) o[v] *= sCorr[qi * 16 + half * 8 + v];
#pragma unroll
    for (int kk = 0; kk < 2; ++kk) {
      v16h a  = load_a_frag(sP + qi * 16 * 72, 72, kk * 32, lane);
      v16h bf = load_b_frag_kn(&sV[ib][0] + kk * 32 * 40 + di * 16, 40, lane);
      o = wmma_f16f32(a, bf, o);
    }
    __syncthreads();       // done reading sK/sV[ib]; may be overwritten next
  }

  // Normalize and stage O (fp32) for the rotation epilogue.
#pragma unroll
  for (int v = 0; v < 8; ++v) {
    const int r = qi * 16 + half * 8 + v;
    sO[r * 36 + di * 16 + col] = o[v] / sSum[r];
  }
  __syncthreads();

  // Inverse (transposed) rotation + write token-major f16 (B, L, 768).
  for (int idx = tid; idx < 64 * 16; idx += 256) {
    const int r = idx >> 4, f = idx & 15;
    const int lTok = qRow0 + r;
    const float x0 = sO[r * 36 + 2 * f];
    const float x1 = sO[r * 36 + 2 * f + 1];
    const float* m4 = mato + ((size_t)lTok * 16 + f) * 4;
    const float y0 = m4[0] * x0 + m4[1] * x1;
    const float y1 = m4[2] * x0 + m4[3] * x1;
    const size_t ob = ((size_t)b * LSEQ + lTok) * EMB + hd * HD;
    outE[ob + 2 * f]     = (_Float16)y0;
    outE[ob + 2 * f + 1] = (_Float16)y1;
  }
}

// ---------------------------------------------------------------------------
// Kernel 3: out = attn @ W_o^T + b_o  (fp32 output).
// 128x128 tile; A tile (f16) staged via async DMA, W_o converted in regs.
// Grid: (NTOK/128, 768/128), block 256.
// ---------------------------------------------------------------------------
__global__ __launch_bounds__(256) void out_proj_kernel(
    const _Float16* __restrict__ A,  // (16384, 768) f16
    const float* __restrict__ Wo,    // (768, 768)
    const float* __restrict__ bo,    // (768)
    float* __restrict__ out)         // (16384, 768) f32
{
  __shared__ __align__(16) _Float16 sA[128 * 40];
  __shared__ __align__(16) _Float16 sB[128 * 40];
  const int tid = threadIdx.x, wave = tid >> 5, lane = tid & 31;
  const int col = lane & 15, half = lane >> 4;
  const int mBlock = blockIdx.x * 128, nBlock = blockIdx.y * 128;
  const int waveM = wave >> 1, waveN = wave & 1;

  v8f acc[2][4] = {};
  for (int kt = 0; kt < EMB; kt += 32) {
    async_stage128_f16(lds_addr32(sA), A + (size_t)mBlock * EMB + kt, EMB, tid);
    stage128_f32cvt(sB, Wo + (size_t)nBlock * EMB + kt, EMB, tid);
    wait_async<0>();
    __syncthreads();
    v16h a0 = load_a_frag(sA + (waveM * 2 + 0) * 16 * 40, 40, 0, lane);
    v16h a1 = load_a_frag(sA + (waveM * 2 + 1) * 16 * 40, 40, 0, lane);
    v16h bf[4];
#pragma unroll
    for (int tn = 0; tn < 4; ++tn)
      bf[tn] = load_b_frag_nk(sB + (waveN * 4 + tn) * 16 * 40, 40, lane);
#pragma unroll
    for (int tn = 0; tn < 4; ++tn) {
      acc[0][tn] = wmma_f16f32(a0, bf[tn], acc[0][tn]);
      acc[1][tn] = wmma_f16f32(a1, bf[tn], acc[1][tn]);
    }
    __syncthreads();
  }
#pragma unroll
  for (int tm = 0; tm < 2; ++tm) {
#pragma unroll
    for (int tn = 0; tn < 4; ++tn) {
      const int n = nBlock + (waveN * 4 + tn) * 16 + col;
      const float bias = bo[n];
#pragma unroll
      for (int v = 0; v < 8; ++v) {
        const int m = mBlock + (waveM * 2 + tm) * 16 + half * 8 + v;
        out[(size_t)m * EMB + n] = acc[tm][tn][v] + bias;
      }
    }
  }
}

// ---------------------------------------------------------------------------
// Launch. Workspace layout (f16):
//   Qr (B,H,L,D) | Kr | Vr | attnOut (B,L,768)  -> 4 * 25,165,824 B ~ 96 MB.
// ---------------------------------------------------------------------------
extern "C" void kernel_launch(void* const* d_in, const int* in_sizes, int n_in,
                              void* d_out, int out_size, void* d_ws,
                              size_t ws_size, hipStream_t stream) {
  (void)in_sizes; (void)n_in; (void)out_size; (void)ws_size;
  const float* x    = (const float*)d_in[0];
  const float* Wqkv = (const float*)d_in[1];
  const float* bqkv = (const float*)d_in[2];
  const float* Wo   = (const float*)d_in[3];
  const float* bo   = (const float*)d_in[4];
  const float* matq = (const float*)d_in[5];
  const float* matk = (const float*)d_in[6];
  const float* matv = (const float*)d_in[7];
  const float* mato = (const float*)d_in[8];

  _Float16* Qr = (_Float16*)d_ws;
  _Float16* Kr = Qr + (size_t)12582912;
  _Float16* Vr = Kr + (size_t)12582912;
  _Float16* attnOut = Vr + (size_t)12582912;
  float* out = (float*)d_out;

  qkv_rot_kernel<<<dim3(NTOK / 128, 2304 / 128), 256, 0, stream>>>(
      x, Wqkv, bqkv, matq, matk, matv, Qr, Kr, Vr);
  attn_kernel<<<dim3(16 * NH * (LSEQ / 64)), 256, 0, stream>>>(
      Qr, Kr, Vr, mato, attnOut);
  out_proj_kernel<<<dim3(NTOK / 128, EMB / 128), 256, 0, stream>>>(
      attnOut, Wo, bo, out);
}
